// FlashLM_v75_10428180594732
// MI455X (gfx1250) — matrix operand order repaired and checked by reading, not compiled, not run on hardware
//
#include <hip/hip_runtime.h>

#define Tt  1024
#define Dd  1024
#define Ll  4
#define FFn 4096
#define Hh  16
#define DHh 64
#define DMm 64
#define Wn  256
#define Vv  32000

typedef __attribute__((ext_vector_type(16))) _Float16 v16h;
typedef __attribute__((ext_vector_type(8)))  float    v8f;

__device__ __forceinline__ v8f wmma_f16(v16h a, v16h b, v8f c) {
  // D = A(16x32 f16) * B(32x16 f16) + C(16x16 f32)
  return __builtin_amdgcn_wmma_f32_16x16x32_f16(false, a, false, b, (short)0, c, false, false);
}

// ---------------- f16-LDS fragment builders (attention kernels) ----------------
// A fragment: 16x32 f16 tile, LDS row-major [row][k], ld in halfs.
__device__ __forceinline__ v16h load_a_frag(const _Float16* lds, int ld, int row0, int k0) {
  int lane = threadIdx.x & 31;
  int m = lane & 15, hl = lane >> 4;
  const _Float16* p = lds + (row0 + m) * ld + k0;
  v16h a;
#pragma unroll
  for (int v = 0; v < 8; ++v) {
    int kb = hl * 16 + 2 * v;
    a[2 * v]     = p[kb];
    a[2 * v + 1] = p[kb + 1];
  }
  return a;
}
// B fragment: 32x16 f16, source stored row-major [n][k].
__device__ __forceinline__ v16h load_b_frag_nk(const _Float16* lds, int ld, int n0, int k0) {
  int lane = threadIdx.x & 31;
  int n = lane & 15, hl = lane >> 4;
  const _Float16* p = lds + (n0 + n) * ld + k0 + hl * 16;
  v16h b;
#pragma unroll
  for (int v = 0; v < 8; ++v) { b[2 * v] = p[2 * v]; b[2 * v + 1] = p[2 * v + 1]; }
  return b;
}
// B fragment from [k][n]-stored tile (e.g. V chunk [key][dh]).
__device__ __forceinline__ v16h load_b_frag_kn(const _Float16* lds, int ld, int k0, int n0) {
  int lane = threadIdx.x & 31;
  int n = (lane & 15) + n0, hl = lane >> 4;
  v16h b;
#pragma unroll
  for (int v = 0; v < 8; ++v) {
    int k = k0 + hl * 16 + 2 * v;
    b[2 * v]     = lds[k * ld + n];
    b[2 * v + 1] = lds[(k + 1) * ld + n];
  }
  return b;
}

// ---------------- fp32-LDS fragment builders (async-staged GEMM) ----------------
__device__ __forceinline__ v16h load_a_frag_f32(const float* lds, int ld, int row0, int k0) {
  int lane = threadIdx.x & 31;
  int m = lane & 15, hl = lane >> 4;
  const float* p = lds + (row0 + m) * ld + k0;
  v16h a;
#pragma unroll
  for (int v = 0; v < 8; ++v) {
    int kb = hl * 16 + 2 * v;
    a[2 * v]     = (_Float16)p[kb];
    a[2 * v + 1] = (_Float16)p[kb + 1];
  }
  return a;
}
__device__ __forceinline__ v16h load_b_frag_f32(const float* lds, int ld, int n0, int k0) {
  int lane = threadIdx.x & 31;
  int n = lane & 15, hl = lane >> 4;
  const float* p = lds + (n0 + n) * ld + k0 + hl * 16;
  v16h b;
#pragma unroll
  for (int v = 0; v < 8; ++v) {
    b[2 * v]     = (_Float16)p[2 * v];
    b[2 * v + 1] = (_Float16)p[2 * v + 1];
  }
  return b;
}

template <int MODE> __device__ __forceinline__ float epi(float x) {
  if (MODE == 1) return 1.f / (1.f + __expf(-x));          // sigmoid
  if (MODE == 2) return x / (1.f + __expf(-x));            // silu
  return x;
}

// Per-lane async 16B copy global -> LDS (ASYNCcnt-tracked, no VGPR data path).
__device__ __forceinline__ void async_copy16(unsigned int lds_byte, const float* gaddr) {
  asm volatile("global_load_async_to_lds_b128 %0, %1, off"
               :: "v"(lds_byte), "v"(gaddr) : "memory");
}

// C[M=1024, N] = act( A[M,K] * W[N,K]^T ).  fp32 in/out, f16 WMMA core.
// Block tile 128x64, 8 waves; wave w -> rows 16w..16w+15, all 64 cols (4 accum tiles).
// Double-buffered LDS tiles filled by GLOBAL_LOAD_ASYNC_TO_LDS_B128; tile t+1 is
// prefetched while tile t is consumed; s_wait_asynccnt 6 drains the older batch
// (each thread issues exactly 6 async ops per tile, async loads complete in order).
template <int MODE>
__global__ void __launch_bounds__(256)
gemm_kernel(const float* __restrict__ A, const float* __restrict__ W,
            float* __restrict__ C, int N, int K) {
  __shared__ __align__(16) float sA[2][128][36];   // 32 K-floats + 4 pad (144B rows)
  __shared__ __align__(16) float sB[2][64][36];
  int bn = blockIdx.x * 64, bm = blockIdx.y * 128;
  int tid = threadIdx.x;
  int wave = tid >> 5;

  unsigned int ldsA[2], ldsB[2];
  ldsA[0] = (unsigned int)(size_t)&sA[0][0][0];
  ldsA[1] = (unsigned int)(size_t)&sA[1][0][0];
  ldsB[0] = (unsigned int)(size_t)&sB[0][0][0];
  ldsB[1] = (unsigned int)(size_t)&sB[1][0][0];

  auto issue_tile = [&](int k0, int p) {
#pragma unroll
    for (int i = 0; i < 4; ++i) {                  // A: 128 rows x 8 x 16B chunks
      int cid = tid + 256 * i;                     // 0..1023
      int row = cid >> 3, ch = cid & 7;
      async_copy16(ldsA[p] + (unsigned int)(row * 144 + ch * 16),
                   A + (size_t)(bm + row) * K + k0 + ch * 4);
    }
#pragma unroll
    for (int i = 0; i < 2; ++i) {                  // W: 64 rows x 8 x 16B chunks
      int cid = tid + 256 * i;                     // 0..511
      int row = cid >> 3, ch = cid & 7;
      int nr = bn + row; if (nr >= N) nr = N - 1;  // clamp; results masked at store
      async_copy16(ldsB[p] + (unsigned int)(row * 144 + ch * 16),
                   W + (size_t)nr * K + k0 + ch * 4);
    }
  };

  v8f acc[4] = {};
  int nt = K >> 5;
  issue_tile(0, 0);
  for (int t = 0; t < nt; ++t) {
    int p = t & 1;
    if (t + 1 < nt) {
      issue_tile((t + 1) * 32, p ^ 1);
      asm volatile("s_wait_asynccnt 0x6" ::: "memory");
    } else {
      asm volatile("s_wait_asynccnt 0x0" ::: "memory");
    }
    __syncthreads();
    v16h a  = load_a_frag_f32(&sA[p][0][0], 36, wave * 16, 0);
    acc[0] = wmma_f16(a, load_b_frag_f32(&sB[p][0][0], 36,  0, 0), acc[0]);
    acc[1] = wmma_f16(a, load_b_frag_f32(&sB[p][0][0], 36, 16, 0), acc[1]);
    acc[2] = wmma_f16(a, load_b_frag_f32(&sB[p][0][0], 36, 32, 0), acc[2]);
    acc[3] = wmma_f16(a, load_b_frag_f32(&sB[p][0][0], 36, 48, 0), acc[3]);
    __syncthreads();
  }
  int lane = tid & 31, nc = lane & 15, hl = lane >> 4;
#pragma unroll
  for (int j = 0; j < 4; ++j) {
#pragma unroll
    for (int r = 0; r < 8; ++r) {
      int m = bm + wave * 16 + hl * 8 + r;
      int n = bn + j * 16 + nc;
      if (n < N) C[(size_t)m * N + n] = epi<MODE>(acc[j][r]);
    }
  }
}

// Sliding-window attention. One wave per (16-query tile, head).
// qkv: [T][3072] (q|k|v, each [H][64]); outc: [T][1024] head-concat.
__global__ void __launch_bounds__(32)
swa_kernel(const float* __restrict__ qkv, float* __restrict__ outc) {
  __shared__ _Float16 sQ[16][72];
  __shared__ _Float16 sKV[32][72];
  __shared__ float    sS[16][328];
  __shared__ _Float16 sP[16][328];
  int q0 = blockIdx.x * 16;
  int h  = blockIdx.y;
  int lane = threadIdx.x & 31;
  int m = lane & 15, hl = lane >> 4;
  {
    int dbase = hl * 32;
    const float* qp = qkv + (size_t)(q0 + m) * 3072 + h * 64 + dbase;
#pragma unroll
    for (int d = 0; d < 32; ++d) sQ[m][dbase + d] = (_Float16)qp[d];
  }
  __syncthreads();
  v16h aq0 = load_a_frag(&sQ[0][0], 72, 0, 0);
  v16h aq1 = load_a_frag(&sQ[0][0], 72, 0, 32);
  int klo = q0 - (Wn - 1); if (klo < 0) klo = 0;
  int kc0 = klo & ~31;
  int nch = (q0 + 16 - kc0 + 31) >> 5;       // <= 10 chunks (<=320 key slots)
  const float scale = 0.125f;                // DH^-0.5
  for (int c = 0; c < nch; ++c) {            // pass 1: scores via WMMA
    int kb = kc0 + c * 32;
    const float* kp = qkv + (size_t)(kb + lane) * 3072 + 1024 + h * 64;
#pragma unroll
    for (int d = 0; d < 64; ++d) sKV[lane][d] = (_Float16)kp[d];
    __syncthreads();
#pragma unroll
    for (int sub = 0; sub < 2; ++sub) {
      v16h bk0 = load_b_frag_nk(&sKV[0][0], 72, sub * 16, 0);
      v16h bk1 = load_b_frag_nk(&sKV[0][0], 72, sub * 16, 32);
      v8f s = {};
      s = wmma_f16(aq0, bk0, s);
      s = wmma_f16(aq1, bk1, s);
      int j = kb + sub * 16 + (lane & 15);
#pragma unroll
      for (int r = 0; r < 8; ++r) {
        int i = q0 + hl * 8 + r;
        int dist = i - j;
        float v = (dist >= 0 && dist < Wn) ? s[r] * scale : -1e30f;
        sS[hl * 8 + r][c * 32 + sub * 16 + (lane & 15)] = v;
      }
    }
    __syncthreads();
  }
  int ncols = nch * 32;                      // pass 2: per-row softmax
  if (lane < 16) {
    float mx = -1e30f;
    for (int cI = 0; cI < ncols; ++cI) mx = fmaxf(mx, sS[lane][cI]);
    float sum = 0.f;
    for (int cI = 0; cI < ncols; ++cI) sum += __expf(sS[lane][cI] - mx);
    float inv = 1.f / sum;
    for (int cI = 0; cI < ncols; ++cI)
      sP[lane][cI] = (_Float16)(__expf(sS[lane][cI] - mx) * inv);
  }
  __syncthreads();
  v8f o0 = {}, o1 = {}, o2 = {}, o3 = {};    // pass 3: O = P @ V via WMMA
  for (int c = 0; c < nch; ++c) {
    int kb = kc0 + c * 32;
    const float* vp = qkv + (size_t)(kb + lane) * 3072 + 2048 + h * 64;
#pragma unroll
    for (int d = 0; d < 64; ++d) sKV[lane][d] = (_Float16)vp[d];
    __syncthreads();
    v16h ap = load_a_frag(&sP[0][0], 328, 0, c * 32);
    o0 = wmma_f16(ap, load_b_frag_kn(&sKV[0][0], 72, 0,  0), o0);
    o1 = wmma_f16(ap, load_b_frag_kn(&sKV[0][0], 72, 0, 16), o1);
    o2 = wmma_f16(ap, load_b_frag_kn(&sKV[0][0], 72, 0, 32), o2);
    o3 = wmma_f16(ap, load_b_frag_kn(&sKV[0][0], 72, 0, 48), o3);
    __syncthreads();
  }
  int nc = lane & 15;
#pragma unroll
  for (int r = 0; r < 8; ++r) {
    int t = q0 + hl * 8 + r;
    float* op = outc + (size_t)t * 1024 + h * 64;
    op[nc]      = o0[r];
    op[16 + nc] = o1[r];
    op[32 + nc] = o2[r];
    op[48 + nc] = o3[r];
  }
}

// Delta memory: out[i] = sum_{j>=i} (q_i.k_j/sqrt(DM))*exp(min(cum_i-cum_j,0)) v_j + beta_i v_i
__global__ void __launch_bounds__(32)
delta_kernel(const float* __restrict__ qn, const float* __restrict__ kn,
             const float* __restrict__ vp, const float* __restrict__ beta,
             const float* __restrict__ cum, float* __restrict__ outc) {
  __shared__ _Float16 sQ[16][72];
  __shared__ _Float16 sKV[32][72];
  __shared__ _Float16 sP[16][40];
  int q0 = blockIdx.x * 16;
  int h  = blockIdx.y;
  int lane = threadIdx.x & 31;
  int m = lane & 15, hl = lane >> 4;
  const float* cumh = cum + (size_t)h * Tt;
  {
    int dbase = hl * 32;
    const float* qp = qn + (size_t)(q0 + m) * 1024 + h * 64 + dbase;
#pragma unroll
    for (int d = 0; d < 32; ++d) sQ[m][dbase + d] = (_Float16)qp[d];
  }
  __syncthreads();
  v16h aq0 = load_a_frag(&sQ[0][0], 72, 0, 0);
  v16h aq1 = load_a_frag(&sQ[0][0], 72, 0, 32);
  const float scale = 0.125f;                // DM^-0.5
  v8f o0 = {}, o1 = {}, o2 = {}, o3 = {};
  for (int kb = q0 & ~31; kb < Tt; kb += 32) {
    const float* kp = kn + (size_t)(kb + lane) * 1024 + h * 64;
#pragma unroll
    for (int d = 0; d < 64; ++d) sKV[lane][d] = (_Float16)kp[d];
    __syncthreads();
#pragma unroll
    for (int sub = 0; sub < 2; ++sub) {
      v16h bk0 = load_b_frag_nk(&sKV[0][0], 72, sub * 16, 0);
      v16h bk1 = load_b_frag_nk(&sKV[0][0], 72, sub * 16, 32);
      v8f s = {};
      s = wmma_f16(aq0, bk0, s);
      s = wmma_f16(aq1, bk1, s);
      int j = kb + sub * 16 + (lane & 15);
      float cj = cumh[j];
#pragma unroll
      for (int r = 0; r < 8; ++r) {
        int i = q0 + hl * 8 + r;
        float w = 0.f;
        if (j >= i) w = __expf(fminf(cumh[i] - cj, 0.f));
        sP[hl * 8 + r][sub * 16 + (lane & 15)] = (_Float16)(s[r] * scale * w);
      }
    }
    __syncthreads();
    const float* vpp = vp + (size_t)(kb + lane) * 1024 + h * 64;
#pragma unroll
    for (int d = 0; d < 64; ++d) sKV[lane][d] = (_Float16)vpp[d];
    __syncthreads();
    v16h ap = load_a_frag(&sP[0][0], 40, 0, 0);
    o0 = wmma_f16(ap, load_b_frag_kn(&sKV[0][0], 72, 0,  0), o0);
    o1 = wmma_f16(ap, load_b_frag_kn(&sKV[0][0], 72, 0, 16), o1);
    o2 = wmma_f16(ap, load_b_frag_kn(&sKV[0][0], 72, 0, 32), o2);
    o3 = wmma_f16(ap, load_b_frag_kn(&sKV[0][0], 72, 0, 48), o3);
    __syncthreads();
  }
  int nc = lane & 15;
#pragma unroll
  for (int r = 0; r < 8; ++r) {
    int t = q0 + hl * 8 + r;
    float b = beta[(size_t)t * Hh + h];
    const float* vr = vp + (size_t)t * 1024 + h * 64;
    float* op = outc + (size_t)t * 1024 + h * 64;
    op[nc]      = o0[r] + b * vr[nc];
    op[16 + nc] = o1[r] + b * vr[16 + nc];
    op[32 + nc] = o2[r] + b * vr[32 + nc];
    op[48 + nc] = o3[r] + b * vr[48 + nc];
  }
}

__global__ void __launch_bounds__(256)
rms_kernel(const float* __restrict__ x, const float* __restrict__ w, float* __restrict__ y) {
  __shared__ float red[8];
  int t = blockIdx.x;
  const float* xr = x + (size_t)t * Dd;
  float s = 0.f;
  for (int d = threadIdx.x; d < Dd; d += 256) { float v = xr[d]; s += v * v; }
#pragma unroll
  for (int o = 16; o > 0; o >>= 1) s += __shfl_xor(s, o, 32);
  if ((threadIdx.x & 31) == 0) red[threadIdx.x >> 5] = s;
  __syncthreads();
  float tot = 0.f;
#pragma unroll
  for (int i = 0; i < 8; ++i) tot += red[i];
  float inv = rsqrtf(tot / Dd + 1e-6f);
  float* yr = y + (size_t)t * Dd;
  for (int d = threadIdx.x; d < Dd; d += 256) yr[d] = xr[d] * inv * w[d];
}

__global__ void __launch_bounds__(256)
embed_rms_kernel(const int* __restrict__ ids, const float* __restrict__ embed,
                 const float* __restrict__ w, float* __restrict__ y) {
  __shared__ float red[8];
  int t = blockIdx.x;
  const float* xr = embed + (size_t)ids[t] * Dd;
  float s = 0.f;
  for (int d = threadIdx.x; d < Dd; d += 256) { float v = xr[d]; s += v * v; }
#pragma unroll
  for (int o = 16; o > 0; o >>= 1) s += __shfl_xor(s, o, 32);
  if ((threadIdx.x & 31) == 0) red[threadIdx.x >> 5] = s;
  __syncthreads();
  float tot = 0.f;
#pragma unroll
  for (int i = 0; i < 8; ++i) tot += red[i];
  float inv = rsqrtf(tot / Dd + 1e-6f);
  float* yr = y + (size_t)t * Dd;
  for (int d = threadIdx.x; d < Dd; d += 256) yr[d] = xr[d] * inv * w[d];
}

__global__ void norm_rows_kernel(float* __restrict__ x) {   // unit-norm each [t][head] 64-vec
  int id = blockIdx.x * blockDim.x + threadIdx.x;
  if (id >= Tt * Hh) return;
  float* p = x + (size_t)id * 64;
  float s = 0.f;
  for (int d = 0; d < 64; ++d) s += p[d] * p[d];
  float inv = 1.f / fmaxf(sqrtf(s), 1e-12f);
  for (int d = 0; d < 64; ++d) p[d] *= inv;
}

__global__ void cumsum_kernel(const float* __restrict__ beta, float* __restrict__ cum) {
  int h = threadIdx.x;
  if (h >= Hh) return;
  float c = 0.f;
  for (int t = 0; t < Tt; ++t) {
    c += __logf(1.f - beta[(size_t)t * Hh + h] + 1e-8f);
    cum[(size_t)h * Tt + t] = c;
  }
}

__global__ void mix_kernel(const float* __restrict__ g, const float* __restrict__ loc,
                           const float* __restrict__ glob, float* __restrict__ dst, int n) {
  int i = blockIdx.x * blockDim.x + threadIdx.x;
  if (i < n) { float gg = g[i]; dst[i] = gg * loc[i] + (1.f - gg) * glob[i]; }
}
__global__ void add_kernel(float* __restrict__ a, const float* __restrict__ b, int n) {
  int i = blockIdx.x * blockDim.x + threadIdx.x;
  if (i < n) a[i] += b[i];
}
__global__ void mul_kernel(float* __restrict__ a, const float* __restrict__ b, int n) {
  int i = blockIdx.x * blockDim.x + threadIdx.x;
  if (i < n) a[i] *= b[i];
}

extern "C" void kernel_launch(void* const* d_in, const int* in_sizes, int n_in,
                              void* d_out, int out_size, void* d_ws, size_t ws_size,
                              hipStream_t stream) {
  (void)in_sizes; (void)n_in; (void)out_size; (void)ws_size;
  const int*   x         = (const int*)  d_in[0];
  const float* embed     = (const float*)d_in[1];
  const float* ln_in_w   = (const float*)d_in[2];
  const float* ln1_w     = (const float*)d_in[3];
  const float* lnd_w     = (const float*)d_in[4];
  const float* qkv_w     = (const float*)d_in[5];
  const float* swa_out_w = (const float*)d_in[6];
  const float* k_w       = (const float*)d_in[7];
  const float* v_w       = (const float*)d_in[8];
  const float* q_w       = (const float*)d_in[9];
  const float* beta_w    = (const float*)d_in[10];
  const float* mem_out_w = (const float*)d_in[11];
  const float* gate_w    = (const float*)d_in[12];
  const float* comb_w    = (const float*)d_in[13];
  const float* ln2_w     = (const float*)d_in[14];
  const float* wg        = (const float*)d_in[15];
  const float* wu        = (const float*)d_in[16];
  const float* wo        = (const float*)d_in[17];
  const float* ln_out_w  = (const float*)d_in[18];
  float* logits = (float*)d_out;

  float* ws = (float*)d_ws;
  const size_t M1 = (size_t)Tt * Dd;       // 1M floats
  float* h    = ws;                        // [T,D]
  float* b1   = ws + 1 * M1;               // h1 / hn
  float* b2   = ws + 2 * M1;               // h2
  float* b3   = ws + 3 * M1;               // qkv (T*3072) / ffg (T*4096)
  float* b4   = ws + 7 * M1;               // k|v|q proj (3M) / ffu (4M)
  float* b5   = ws + 11 * M1;              // attention concat
  float* bA   = ws + 12 * M1;              // local
  float* bB   = ws + 13 * M1;              // global ctx / ffn out
  float* bC   = ws + 14 * M1;              // gate / combine proj
  float* beta = ws + 15 * M1;              // [T,H]
  float* cum  = beta + (size_t)Tt * Hh;    // [H,T]

  embed_rms_kernel<<<Tt, 256, 0, stream>>>(x, embed, ln_in_w, h);

  for (int l = 0; l < Ll; ++l) {
    rms_kernel<<<Tt, 256, 0, stream>>>(h, ln1_w + (size_t)l * Dd, b1);
    rms_kernel<<<Tt, 256, 0, stream>>>(h, lnd_w + (size_t)l * Dd, b2);

    // ---- sliding-window branch ----
    gemm_kernel<0><<<dim3(3072 / 64, Tt / 128), 256, 0, stream>>>(
        b1, qkv_w + (size_t)l * 3072 * Dd, b3, 3072, Dd);
    swa_kernel<<<dim3(Tt / 16, Hh), 32, 0, stream>>>(b3, b5);
    gemm_kernel<0><<<dim3(Dd / 64, Tt / 128), 256, 0, stream>>>(
        b5, swa_out_w + (size_t)l * Dd * Dd, bA, Dd, Dd);

    // ---- delta memory branch ----
    float* kp = b4; float* vp = b4 + M1; float* qp = b4 + 2 * M1;
    gemm_kernel<0><<<dim3(Dd / 64, Tt / 128), 256, 0, stream>>>(
        b2, k_w + (size_t)l * Dd * Dd, kp, Dd, Dd);
    gemm_kernel<0><<<dim3(Dd / 64, Tt / 128), 256, 0, stream>>>(
        b2, v_w + (size_t)l * Dd * Dd, vp, Dd, Dd);
    gemm_kernel<0><<<dim3(Dd / 64, Tt / 128), 256, 0, stream>>>(
        b2, q_w + (size_t)l * Dd * Dd, qp, Dd, Dd);
    gemm_kernel<1><<<dim3(1, Tt / 128), 256, 0, stream>>>(
        b2, beta_w + (size_t)l * Hh * Dd, beta, Hh, Dd);
    norm_rows_kernel<<<(Tt * Hh + 255) / 256, 256, 0, stream>>>(kp);
    norm_rows_kernel<<<(Tt * Hh + 255) / 256, 256, 0, stream>>>(qp);
    cumsum_kernel<<<1, 32, 0, stream>>>(beta, cum);
    delta_kernel<<<dim3(Tt / 16, Hh), 32, 0, stream>>>(qp, kp, vp, beta, cum, b5);
    gemm_kernel<0><<<dim3(Dd / 64, Tt / 128), 256, 0, stream>>>(
        b5, mem_out_w + (size_t)l * Dd * Dd, bB, Dd, Dd);

    // ---- gate / combine / residual ----
    gemm_kernel<1><<<dim3(Dd / 64, Tt / 128), 256, 0, stream>>>(
        b1, gate_w + (size_t)l * Dd * Dd, bC, Dd, Dd);
    mix_kernel<<<(int)(M1 / 256), 256, 0, stream>>>(bC, bA, bB, bA, (int)M1);
    gemm_kernel<0><<<dim3(Dd / 64, Tt / 128), 256, 0, stream>>>(
        bA, comb_w + (size_t)l * Dd * Dd, bC, Dd, Dd);
    add_kernel<<<(int)(M1 / 256), 256, 0, stream>>>(h, bC, (int)M1);

    // ---- FFN ----
    rms_kernel<<<Tt, 256, 0, stream>>>(h, ln2_w + (size_t)l * Dd, b1);
    gemm_kernel<2><<<dim3(FFn / 64, Tt / 128), 256, 0, stream>>>(
        b1, wg + (size_t)l * FFn * Dd, b3, FFn, Dd);
    gemm_kernel<0><<<dim3(FFn / 64, Tt / 128), 256, 0, stream>>>(
        b1, wu + (size_t)l * FFn * Dd, b4, FFn, Dd);
    mul_kernel<<<(int)((size_t)Tt * FFn / 256), 256, 0, stream>>>(b3, b4, Tt * FFn);
    gemm_kernel<0><<<dim3(Dd / 64, Tt / 128), 256, 0, stream>>>(
        b3, wo + (size_t)l * Dd * FFn, bB, Dd, FFn);
    add_kernel<<<(int)(M1 / 256), 256, 0, stream>>>(h, bB, (int)M1);
  }

  // ---- tied head ----
  rms_kernel<<<Tt, 256, 0, stream>>>(h, ln_out_w, b1);
  gemm_kernel<0><<<dim3(Vv / 64, Tt / 128), 256, 0, stream>>>(b1, embed, logits, Vv, Dd);
}
